// MSA_1305670058295
// MI455X (gfx1250) — matrix operand order repaired and checked by reading, not compiled, not run on hardware
//
#include <hip/hip_runtime.h>
#include <hip/hip_bf16.h>

// MI455X / gfx1250 multi-head self-attention (B=2, S=4096, D=768, H=12, dk=64)
// - all matmuls via v_wmma_f32_16x16x32_bf16 (bf16 in, fp32 accum)
// - flash-style online softmax (scores never hit HBM)
// - LDS tiles filled with global_load_async_to_lds_b128 (ASYNCcnt path),
//   double-buffered so DMA overlaps WMMA compute; one barrier per iteration.

typedef __bf16 bf16;
typedef __attribute__((ext_vector_type(16))) __bf16 v16bf;
typedef __attribute__((ext_vector_type(8)))  __bf16 v8bf;
typedef __attribute__((ext_vector_type(8)))  float  v8f;

#define D_MODEL 768
#define SEQ     4096
#define BATCH   2
#define NHEAD   12
#define DK      64
#define ROWS    (BATCH * SEQ)     // 8192

static __device__ __forceinline__ v16bf cat8(v8bf lo, v8bf hi) {
    return __builtin_shufflevector(lo, hi, 0,1,2,3,4,5,6,7,8,9,10,11,12,13,14,15);
}

static __device__ __forceinline__ v8f wmma_bf16(v16bf a, v16bf b, v8f c) {
    return __builtin_amdgcn_wmma_f32_16x16x32_bf16(false, a, false, b, (short)0, c,
                                                   false, false);
}

// Async global -> LDS copy, 16B per lane (CDNA5 ASYNCcnt path, no VGPR round trip).
// LDS offset = low 32 bits of the generic pointer (flat LDS aperture: addr[31:0]).
static __device__ __forceinline__ void async_copy_b128(void* lds_ptr, const void* gptr) {
    const unsigned lds_off = (unsigned)(uintptr_t)lds_ptr;
    asm volatile("global_load_async_to_lds_b128 %0, %1, off"
                 :: "v"(lds_off), "v"(gptr) : "memory");
}
static __device__ __forceinline__ void wait_async() {
    asm volatile("s_wait_asynccnt 0" ::: "memory");
}

// ---------------------------------------------------------------------------
// Prep: fp32 -> bf16 conversion of x
// ---------------------------------------------------------------------------
__global__ void k_cvt_bf16(const float* __restrict__ in, bf16* __restrict__ out, int n) {
    int i = (blockIdx.x * blockDim.x + threadIdx.x) * 4;
    if (i + 3 < n) {
        float4 v = *(const float4*)(in + i);
        out[i + 0] = (bf16)v.x;
        out[i + 1] = (bf16)v.y;
        out[i + 2] = (bf16)v.z;
        out[i + 3] = (bf16)v.w;
    }
}

// Prep: W[k][n] (768x768 fp32) -> Wt[n][k] bf16 (WMMA B-frag reads K-contiguous)
__global__ void k_transpose_w(const float* __restrict__ W, bf16* __restrict__ Wt) {
    __shared__ float tile[16][17];
    const int bx = blockIdx.x * 16, by = blockIdx.y * 16;
    const int tx = threadIdx.x & 15, ty = threadIdx.x >> 4;   // 256 threads
    tile[ty][tx] = W[(size_t)(by + ty) * D_MODEL + bx + tx];
    __syncthreads();
    Wt[(size_t)(bx + ty) * D_MODEL + by + tx] = (bf16)tile[tx][ty];
}

// ---------------------------------------------------------------------------
// Shared helper: issue async fill of one 64x32 bf16 GEMM tile pair
// ---------------------------------------------------------------------------
static __device__ __forceinline__ void gemm_fill(bf16 (*As)[40], bf16 (*Bs)[40],
                                                 const bf16* __restrict__ X,
                                                 const bf16* __restrict__ Wt,
                                                 int m0, int n0, int k0, int t) {
#pragma unroll
    for (int it = 0; it < 2; ++it) {
        const int c = it * 128 + t;               // 0..255
        const int r = c >> 2, s8 = (c & 3) * 8;
        async_copy_b128(&As[r][s8], &X [(size_t)(m0 + r) * D_MODEL + k0 + s8]);
        async_copy_b128(&Bs[r][s8], &Wt[(size_t)(n0 + r) * D_MODEL + k0 + s8]);
    }
}

// ---------------------------------------------------------------------------
// Fused QKV projection: X[8192x768] bf16 @ Wt[2304(n)][768(k)] bf16 + bias
// Q,K -> [B, H, S, 64] bf16;  V -> transposed [B, H, 64, S] bf16.
// ---------------------------------------------------------------------------
__global__ void k_qkv_gemm(const bf16* __restrict__ X, const bf16* __restrict__ Wt,
                           const float* __restrict__ bq, const float* __restrict__ bk,
                           const float* __restrict__ bv,
                           bf16* __restrict__ Qg, bf16* __restrict__ Kg,
                           bf16* __restrict__ Vg) {
    __shared__ bf16 As[2][64][40];   // ping-pong [m][k]
    __shared__ bf16 Bs[2][64][40];   // ping-pong [n][k]
    const int t = threadIdx.x;                    // 128 threads = 4 waves
    const int m0 = blockIdx.x * 64;
    const int n0 = blockIdx.y * 64;
    const int lane = t & 31, w = t >> 5;
    const int lo = lane & 15, hi = lane >> 4;

    v8f acc[4] = {};

    gemm_fill(As[0], Bs[0], X, Wt, m0, n0, 0, t);
    wait_async();
    __syncthreads();

    const int NKT = D_MODEL / 32;                 // 24
    for (int ki = 0; ki < NKT; ++ki) {
        const int cur = ki & 1;
        if (ki + 1 < NKT)
            gemm_fill(As[cur ^ 1], Bs[cur ^ 1], X, Wt, m0, n0, (ki + 1) * 32, t);

        const v16bf a = cat8(*(const v8bf*)&As[cur][w * 16 + lo][hi * 16],
                             *(const v8bf*)&As[cur][w * 16 + lo][hi * 16 + 8]);
#pragma unroll
        for (int nt = 0; nt < 4; ++nt) {
            const v16bf b = cat8(*(const v8bf*)&Bs[cur][nt * 16 + lo][hi * 16],
                                 *(const v8bf*)&Bs[cur][nt * 16 + lo][hi * 16 + 8]);
            acc[nt] = wmma_bf16(a, b, acc[nt]);
        }
        wait_async();          // our next-tile DMA has landed
        __syncthreads();       // everyone done reading cur + everyone's DMA landed
    }

    // Epilogue: bias + scatter (Q/K row-major per head, V transposed per head)
    const int which = n0 / D_MODEL;
    const int colbase = n0 % D_MODEL;
    const float* bias = (which == 0) ? bq : (which == 1) ? bk : bv;
    bf16* dst = (which == 0) ? Qg : (which == 1) ? Kg : Vg;
#pragma unroll
    for (int nt = 0; nt < 4; ++nt) {
        const int col = colbase + nt * 16 + lo;
        const int h = col >> 6, dk = col & 63;
        const float bb = bias[col];
#pragma unroll
        for (int i = 0; i < 8; ++i) {
            const int m = m0 + w * 16 + hi * 8 + i;       // C-frag row
            const int b_ = m >> 12, s = m & 4095;
            const size_t head = (size_t)b_ * NHEAD + h;
            const size_t idx = (which == 2)
                ? (head * DK + dk) * SEQ + s      // V^T: [B,H,dk,S]
                : (head * SEQ + s) * DK + dk;     // Q,K: [B,H,S,dk]
            dst[idx] = (bf16)(acc[nt][i] + bb);
        }
    }
}

// ---------------------------------------------------------------------------
// Flash attention: one block per (b*H+h, 64-query tile); 4 waves x 16 q-rows.
// Double-buffered K/V^T tiles; 16 WMMAs per 64-key tile.
// ---------------------------------------------------------------------------
static __device__ __forceinline__ void attn_fill(bf16 (*Ksb)[72], bf16 (*Vtb)[72],
                                                 const bf16* __restrict__ Kg,
                                                 const bf16* __restrict__ Vgt,
                                                 size_t headBase, size_t vBase,
                                                 int kt, int t) {
#pragma unroll
    for (int it = 0; it < 4; ++it) {
        const int c = it * 128 + t;               // 0..511
        const int r = c >> 3, s8 = (c & 7) * 8;
        async_copy_b128(&Ksb[r][s8], &Kg [(headBase + kt * 64 + r) * DK + s8]);
        async_copy_b128(&Vtb[r][s8], &Vgt[vBase + (size_t)r * SEQ + kt * 64 + s8]);
    }
}

__global__ void k_attn(const bf16* __restrict__ Qg, const bf16* __restrict__ Kg,
                       const bf16* __restrict__ Vgt, bf16* __restrict__ Ctx) {
    __shared__ bf16 Ks[2][64][72];     // [key][d]   row-major: QK^T B-frag contiguous
    __shared__ bf16 Vt[2][64][72];     // [d][key]   from V^T:  P.V  B-frag contiguous
    __shared__ bf16 Ps[4][16][72];     // per-wave P staging (C-layout -> A-layout)

    const int t = threadIdx.x;                    // 128 threads = 4 waves
    const int y = blockIdx.y;                     // b*NHEAD + h
    const int q0 = blockIdx.x * 64;
    const int lane = t & 31, w = t >> 5;
    const int lo = lane & 15, hi = lane >> 4;
    const size_t headBase = (size_t)y * SEQ;      // rows of Q/K for this head
    const size_t vBase = (size_t)y * DK * SEQ;    // base of V^T for this head

    // Q A-fragments for this wave's 16 rows (d split into two 32-wide K windows)
    const bf16* qrow = Qg + (headBase + q0 + w * 16 + lo) * DK;
    v16bf qa[2];
#pragma unroll
    for (int c = 0; c < 2; ++c)
        qa[c] = cat8(*(const v8bf*)&qrow[c * 32 + hi * 16],
                     *(const v8bf*)&qrow[c * 32 + hi * 16 + 8]);

    v8f acc[4] = {};
    float m_run[8], l_run[8];
#pragma unroll
    for (int i = 0; i < 8; ++i) { m_run[i] = -3.0e38f; l_run[i] = 0.0f; }

    attn_fill(Ks[0], Vt[0], Kg, Vgt, headBase, vBase, 0, t);
    wait_async();
    __syncthreads();

    const int NT = SEQ / 64;                      // 64 key tiles
    for (int kt = 0; kt < NT; ++kt) {
        const int cur = kt & 1;
        if (kt + 1 < NT)
            attn_fill(Ks[cur ^ 1], Vt[cur ^ 1], Kg, Vgt, headBase, vBase, kt + 1, t);
        if (kt + 2 < NT)
            __builtin_prefetch(&Kg[(headBase + (kt + 2) * 64 + (t >> 1)) * DK], 0, 1);

        // S = Q K^T (scaled)
        v8f st[4] = {};
#pragma unroll
        for (int c = 0; c < 2; ++c) {
#pragma unroll
            for (int nt = 0; nt < 4; ++nt) {
                const v16bf b =
                    cat8(*(const v8bf*)&Ks[cur][nt * 16 + lo][c * 32 + hi * 16],
                         *(const v8bf*)&Ks[cur][nt * 16 + lo][c * 32 + hi * 16 + 8]);
                st[nt] = wmma_bf16(qa[c], b, st[nt]);
            }
        }
#pragma unroll
        for (int nt = 0; nt < 4; ++nt)
#pragma unroll
            for (int i = 0; i < 8; ++i) st[nt][i] *= 0.125f;   // 1/sqrt(64)

        // Row-wise online softmax. C-frag: row = hi*8+i, col = nt*16+lo.
        // xor 1/2/4/8 reduces across the 16 lanes holding one row.
        float mt[8], alpha[8], rs[8];
#pragma unroll
        for (int i = 0; i < 8; ++i) {
            mt[i] = fmaxf(fmaxf(st[0][i], st[1][i]), fmaxf(st[2][i], st[3][i]));
#pragma unroll
            for (int d = 1; d <= 8; d <<= 1) mt[i] = fmaxf(mt[i], __shfl_xor(mt[i], d, 32));
            const float mn = fmaxf(m_run[i], mt[i]);
            alpha[i] = exp2f((m_run[i] - mn) * 1.44269504f);
            m_run[i] = mn;
            rs[i] = 0.0f;
        }
#pragma unroll
        for (int nt = 0; nt < 4; ++nt)
#pragma unroll
            for (int i = 0; i < 8; ++i) {
                const float p = exp2f((st[nt][i] - m_run[i]) * 1.44269504f);
                st[nt][i] = p;
                rs[i] += p;
            }
#pragma unroll
        for (int i = 0; i < 8; ++i) {
#pragma unroll
            for (int d = 1; d <= 8; d <<= 1) rs[i] += __shfl_xor(rs[i], d, 32);
            l_run[i] = l_run[i] * alpha[i] + rs[i];
        }
#pragma unroll
        for (int nt = 0; nt < 4; ++nt)
#pragma unroll
            for (int i = 0; i < 8; ++i) acc[nt][i] *= alpha[i];

        // Stage P through per-wave LDS (C-layout -> A-layout). Same-wave LDS is in-order.
#pragma unroll
        for (int nt = 0; nt < 4; ++nt)
#pragma unroll
            for (int i = 0; i < 8; ++i)
                Ps[w][hi * 8 + i][nt * 16 + lo] = (bf16)st[nt][i];
        __builtin_amdgcn_wave_barrier();

        // acc += P V
#pragma unroll
        for (int c = 0; c < 2; ++c) {
            const v16bf pa = cat8(*(const v8bf*)&Ps[w][lo][c * 32 + hi * 16],
                                  *(const v8bf*)&Ps[w][lo][c * 32 + hi * 16 + 8]);
#pragma unroll
            for (int nt = 0; nt < 4; ++nt) {
                const v16bf vb =
                    cat8(*(const v8bf*)&Vt[cur][nt * 16 + lo][c * 32 + hi * 16],
                         *(const v8bf*)&Vt[cur][nt * 16 + lo][c * 32 + hi * 16 + 8]);
                acc[nt] = wmma_bf16(pa, vb, acc[nt]);
            }
        }
        wait_async();          // next-tile DMA landed (overlapped with the above)
        __syncthreads();       // all waves done reading cur
    }

    // ctx[b][s][h*64 + d] bf16
    const int b_ = y / NHEAD, h = y % NHEAD;
#pragma unroll
    for (int i = 0; i < 8; ++i) {
        const float inv = 1.0f / l_run[i];
        const int q = q0 + w * 16 + hi * 8 + i;
        bf16* crow = Ctx + ((size_t)b_ * SEQ + q) * D_MODEL + h * DK;
#pragma unroll
        for (int nt = 0; nt < 4; ++nt)
            crow[nt * 16 + lo] = (bf16)(acc[nt][i] * inv);
    }
}

// ---------------------------------------------------------------------------
// Output projection: ctx[8192x768] bf16 @ Wo + bo -> fp32 out
// ---------------------------------------------------------------------------
__global__ void k_out_gemm(const bf16* __restrict__ Xc, const bf16* __restrict__ Wot,
                           const float* __restrict__ bo, float* __restrict__ Out) {
    __shared__ bf16 As[2][64][40];
    __shared__ bf16 Bs[2][64][40];
    const int t = threadIdx.x;
    const int m0 = blockIdx.x * 64;
    const int n0 = blockIdx.y * 64;
    const int lane = t & 31, w = t >> 5;
    const int lo = lane & 15, hi = lane >> 4;

    v8f acc[4] = {};

    gemm_fill(As[0], Bs[0], Xc, Wot, m0, n0, 0, t);
    wait_async();
    __syncthreads();

    const int NKT = D_MODEL / 32;
    for (int ki = 0; ki < NKT; ++ki) {
        const int cur = ki & 1;
        if (ki + 1 < NKT)
            gemm_fill(As[cur ^ 1], Bs[cur ^ 1], Xc, Wot, m0, n0, (ki + 1) * 32, t);

        const v16bf a = cat8(*(const v8bf*)&As[cur][w * 16 + lo][hi * 16],
                             *(const v8bf*)&As[cur][w * 16 + lo][hi * 16 + 8]);
#pragma unroll
        for (int nt = 0; nt < 4; ++nt) {
            const v16bf b = cat8(*(const v8bf*)&Bs[cur][nt * 16 + lo][hi * 16],
                                 *(const v8bf*)&Bs[cur][nt * 16 + lo][hi * 16 + 8]);
            acc[nt] = wmma_bf16(a, b, acc[nt]);
        }
        wait_async();
        __syncthreads();
    }

#pragma unroll
    for (int nt = 0; nt < 4; ++nt) {
        const int n = n0 + nt * 16 + lo;
        const float bb = bo[n];
#pragma unroll
        for (int i = 0; i < 8; ++i) {
            const int m = m0 + w * 16 + hi * 8 + i;
            Out[(size_t)m * D_MODEL + n] = acc[nt][i] + bb;
        }
    }
}

// ---------------------------------------------------------------------------
extern "C" void kernel_launch(void* const* d_in, const int* in_sizes, int n_in,
                              void* d_out, int out_size, void* d_ws, size_t ws_size,
                              hipStream_t stream) {
    const float* x  = (const float*)d_in[0];
    const float* Wq = (const float*)d_in[1];
    const float* bq = (const float*)d_in[2];
    const float* Wk = (const float*)d_in[3];
    const float* bk = (const float*)d_in[4];
    const float* Wv = (const float*)d_in[5];
    const float* bv = (const float*)d_in[6];
    const float* Wo = (const float*)d_in[7];
    const float* bo = (const float*)d_in[8];
    float* out = (float*)d_out;

    // Workspace layout (all offsets 256B-aligned), total ~67.6 MB
    char* ws = (char*)d_ws;
    const size_t NX = (size_t)ROWS * D_MODEL;      // 6,291,456 elems
    const size_t NW = (size_t)D_MODEL * D_MODEL;   //   589,824 elems
    size_t off = 0;
    bf16* xb  = (bf16*)(ws + off); off += NX * 2;
    bf16* wt  = (bf16*)(ws + off); off += 3 * NW * 2;   // Wq^T | Wk^T | Wv^T
    bf16* wot = (bf16*)(ws + off); off += NW * 2;
    bf16* Qg  = (bf16*)(ws + off); off += NX * 2;
    bf16* Kg  = (bf16*)(ws + off); off += NX * 2;
    bf16* Vg  = (bf16*)(ws + off); off += NX * 2;       // V^T: [B,H,dk,S]
    bf16* ctx = (bf16*)(ws + off); off += NX * 2;
    (void)ws_size; (void)in_sizes; (void)n_in; (void)out_size;

    k_cvt_bf16<<<dim3((unsigned)(NX / 1024)), 256, 0, stream>>>(x, xb, (int)NX);
    const dim3 tg(48, 48);
    k_transpose_w<<<tg, 256, 0, stream>>>(Wq, wt);
    k_transpose_w<<<tg, 256, 0, stream>>>(Wk, wt + NW);
    k_transpose_w<<<tg, 256, 0, stream>>>(Wv, wt + 2 * NW);
    k_transpose_w<<<tg, 256, 0, stream>>>(Wo, wot);

    k_qkv_gemm<<<dim3(ROWS / 64, (3 * D_MODEL) / 64), 128, 0, stream>>>(
        xb, wt, bq, bk, bv, Qg, Kg, Vg);

    k_attn<<<dim3(SEQ / 64, BATCH * NHEAD), 128, 0, stream>>>(Qg, Kg, Vg, ctx);

    k_out_gemm<<<dim3(ROWS / 64, D_MODEL / 64), 128, 0, stream>>>(ctx, wot, bo, out);
}
